// OrbMessagePassingLayer_15693810499874
// MI455X (gfx1250) — compile-verified, matrix operand order checked by reading, not run on hardware
//
#include <hip/hip_runtime.h>
#include <hip/hip_bf16.h>

// ---------------------------------------------------------------------------
// OrbMessagePassingLayer for MI455X (gfx1250), wave32 + WMMA bf16.
// M-tile = 64 rows/block for 4x register-level reuse of weight fragments.
// ---------------------------------------------------------------------------

#define NN 40000
#define EE 640000
#define CC 128
#define HH 256
#define K0 384              // 3*C
#define MT 64               // rows (edges/nodes) per block
#define CUTOFF_INV (1.0f/6.0f)
#define LN_EPS 1e-5f

typedef __attribute__((ext_vector_type(16))) __bf16 v16bf;
typedef __attribute__((ext_vector_type(8)))  __bf16 v8bf;
typedef __attribute__((ext_vector_type(8)))  float  v8f;

// LDS layout (bytes), sOut aliases sEF (dead after layer 0, barrier between):
//   sEF : 64*384*2 = 49152   @ 0
//   sH0 : 64*256*2 = 32768   @ 49152
//   sH1 : 64*256*2 = 32768   @ 81920
//   sOut: 64*128*4 = 32768   @ 0 (alias of sEF)
#define SMEM_BYTES (49152 + 32768 + 32768)

// Load a 16x32 bf16 WMMA fragment (A or B) per ISA 7.12.2 layout:
// lane holds 8 elems at [k .. k+7] and 8 elems at [k+16 .. k+23],
// where k = k0 + (lane<16 ? 0 : 8).  Two 16-byte loads.
__device__ __forceinline__ v16bf ldfrag(const __bf16* p) {
    v16bf f;
    ((v8bf*)&f)[0] = *(const v8bf*)(p);
    ((v8bf*)&f)[1] = *(const v8bf*)(p + 16);
    return f;
}

__device__ __forceinline__ float siluf(float x) {
    return x / (1.0f + __expf(-x));
}

__device__ __forceinline__ float atomicMaxF(float* addr, float val) {
    unsigned int* ua = (unsigned int*)addr;
    unsigned int old = *ua;
    while (true) {
        float f = __uint_as_float(old);
        if (f >= val) break;
        unsigned int assumed = old;
        old = atomicCAS(ua, assumed, __float_as_uint(val));
        if (old == assumed) break;
    }
    return __uint_as_float(old);
}

// ---------------------------------------------------------------------------
// Shared 3-layer MLP body on a 64-row tile living in LDS (bf16).
// Each wave owns a 16-wide output column tile and iterates 4 M row-tiles,
// holding 4 f32 accumulator tiles so every B fragment is reused 4x.
// Weights WTx are TRANSPOSED bf16 [Nout][K] so B-fragments are contiguous.
// Called by all 256 threads (8 waves).
// ---------------------------------------------------------------------------
__device__ __forceinline__ void mlp3_body(
    const __bf16* __restrict__ sEF, __bf16* __restrict__ sH0,
    __bf16* __restrict__ sH1, float* __restrict__ sOut,
    const __bf16* __restrict__ WT0, const float* __restrict__ B0,
    const __bf16* __restrict__ WT1, const float* __restrict__ B1,
    const __bf16* __restrict__ WT2, const float* __restrict__ B2,
    int lane, int wave)
{
    const int lm = lane & 15;        // row (A) / col (B,D) within tile
    const int hi = lane >> 4;        // lane-group
    const int kb = hi * 8;           // K sub-offset for 16-bit fragments

    // ---- layer0: [64,384] @ [384,256] + b0, SiLU -> sH0 ------------------
    for (int i = 0; i < 2; ++i) {
        const int col = (wave * 2 + i) * 16 + lm;
        const float bv = B0[col];
        v8f acc[4];
        #pragma unroll
        for (int m = 0; m < 4; ++m)
            #pragma unroll
            for (int r = 0; r < 8; ++r) acc[m][r] = bv;
        #pragma unroll
        for (int kt = 0; kt < K0 / 32; ++kt) {
            const int k0 = kt * 32;
            v16bf b = ldfrag(WT0 + (size_t)col * K0 + k0 + kb);
            #pragma unroll
            for (int m = 0; m < 4; ++m) {
                v16bf a = ldfrag(sEF + (m * 16 + lm) * K0 + k0 + kb);
                acc[m] = __builtin_amdgcn_wmma_f32_16x16x32_bf16(
                    false, a, false, b, (short)0, acc[m], false, false);
            }
        }
        #pragma unroll
        for (int m = 0; m < 4; ++m)
            #pragma unroll
            for (int r = 0; r < 8; ++r)
                sH0[(m * 16 + r + hi * 8) * HH + col] = (__bf16)siluf(acc[m][r]);
    }
    __syncthreads();

    // ---- layer1: [64,256] @ [256,256] + b1, SiLU -> sH1 ------------------
    for (int i = 0; i < 2; ++i) {
        const int col = (wave * 2 + i) * 16 + lm;
        const float bv = B1[col];
        v8f acc[4];
        #pragma unroll
        for (int m = 0; m < 4; ++m)
            #pragma unroll
            for (int r = 0; r < 8; ++r) acc[m][r] = bv;
        #pragma unroll
        for (int kt = 0; kt < HH / 32; ++kt) {
            const int k0 = kt * 32;
            v16bf b = ldfrag(WT1 + (size_t)col * HH + k0 + kb);
            #pragma unroll
            for (int m = 0; m < 4; ++m) {
                v16bf a = ldfrag(sH0 + (m * 16 + lm) * HH + k0 + kb);
                acc[m] = __builtin_amdgcn_wmma_f32_16x16x32_bf16(
                    false, a, false, b, (short)0, acc[m], false, false);
            }
        }
        #pragma unroll
        for (int m = 0; m < 4; ++m)
            #pragma unroll
            for (int r = 0; r < 8; ++r)
                sH1[(m * 16 + r + hi * 8) * HH + col] = (__bf16)siluf(acc[m][r]);
    }
    __syncthreads();

    // ---- layer2: [64,256] @ [256,128] + b2 -> sOut (f32) -----------------
    {
        const int col = wave * 16 + lm;          // 8 waves cover 128 cols
        const float bv = B2[col];
        v8f acc[4];
        #pragma unroll
        for (int m = 0; m < 4; ++m)
            #pragma unroll
            for (int r = 0; r < 8; ++r) acc[m][r] = bv;
        #pragma unroll
        for (int kt = 0; kt < HH / 32; ++kt) {
            const int k0 = kt * 32;
            v16bf b = ldfrag(WT2 + (size_t)col * HH + k0 + kb);
            #pragma unroll
            for (int m = 0; m < 4; ++m) {
                v16bf a = ldfrag(sH1 + (m * 16 + lm) * HH + k0 + kb);
                acc[m] = __builtin_amdgcn_wmma_f32_16x16x32_bf16(
                    false, a, false, b, (short)0, acc[m], false, false);
            }
        }
        #pragma unroll
        for (int m = 0; m < 4; ++m)
            #pragma unroll
            for (int r = 0; r < 8; ++r)
                sOut[(m * 16 + r + hi * 8) * CC + col] = acc[m][r];
    }
    __syncthreads();
}

// ---------------------------------------------------------------------------
// Kernel 1: init workspace (maxes=-inf, sums=0, scatter accumulators=0)
// ---------------------------------------------------------------------------
__global__ __launch_bounds__(256) void init_kernel(
    float* rMax, float* sMax, float* rSum, float* sSum,
    float* sentA, float* recvA)
{
    const int t = blockIdx.x * 256 + threadIdx.x;   // grid covers N*C
    sentA[t] = 0.0f;
    recvA[t] = 0.0f;
    if (t < NN) {
        rMax[t] = -__builtin_inff();
        sMax[t] = -__builtin_inff();
        rSum[t] = 0.0f;
        sSum[t] = 0.0f;
    }
}

// ---------------------------------------------------------------------------
// Kernel 2: transpose f32 weight [K][Nn] -> bf16 WT [Nn][K]
// ---------------------------------------------------------------------------
__global__ __launch_bounds__(256) void transpose_kernel(
    const float* __restrict__ W, __bf16* __restrict__ WT, int K, int Nn)
{
    const int t = blockIdx.x * 256 + threadIdx.x;
    if (t >= K * Nn) return;
    const int k = t / Nn, n = t - k * Nn;
    WT[(size_t)n * K + k] = (__bf16)W[t];
}

// ---------------------------------------------------------------------------
// Kernel 3: attention logits + per-segment max
// ---------------------------------------------------------------------------
__global__ __launch_bounds__(256) void logits_kernel(
    const float* __restrict__ edge_emb,
    const float* __restrict__ recv_w, const float* __restrict__ recv_b,
    const float* __restrict__ send_w, const float* __restrict__ send_b,
    const int* __restrict__ senders, const int* __restrict__ receivers,
    float* __restrict__ rLog, float* __restrict__ sLog,
    float* __restrict__ rMax, float* __restrict__ sMax)
{
    const int e = blockIdx.x * 256 + threadIdx.x;   // E divisible by 256
    const float4* row = (const float4*)(edge_emb + (size_t)e * CC);
    const float4* wr = (const float4*)recv_w;
    const float4* ws = (const float4*)send_w;
    float ar = 0.0f, as = 0.0f;
    #pragma unroll 8
    for (int i = 0; i < CC / 4; ++i) {
        float4 v = row[i], a = wr[i], b = ws[i];
        ar += v.x * a.x + v.y * a.y + v.z * a.z + v.w * a.w;
        as += v.x * b.x + v.y * b.y + v.z * b.z + v.w * b.w;
    }
    ar += recv_b[0];
    as += send_b[0];
    rLog[e] = ar;
    sLog[e] = as;
    atomicMaxF(&rMax[receivers[e]], ar);
    atomicMaxF(&sMax[senders[e]], as);
}

// ---------------------------------------------------------------------------
// Kernel 4: exp(logit - max) + per-segment sum (in-place on logit arrays)
// ---------------------------------------------------------------------------
__global__ __launch_bounds__(256) void expsum_kernel(
    const int* __restrict__ senders, const int* __restrict__ receivers,
    float* __restrict__ rLog, float* __restrict__ sLog,
    const float* __restrict__ rMax, const float* __restrict__ sMax,
    float* __restrict__ rSum, float* __restrict__ sSum)
{
    const int e = blockIdx.x * 256 + threadIdx.x;
    const int s = senders[e], r = receivers[e];
    const float re = __expf(rLog[e] - rMax[r]);
    const float se = __expf(sLog[e] - sMax[s]);
    rLog[e] = re;
    sLog[e] = se;
    atomicAdd(&rSum[r], re);
    atomicAdd(&sSum[s], se);
}

// ---------------------------------------------------------------------------
// Kernel 5: edge MLP + layernorm + residual + weighted scatter
// One block = 64 edges, 256 threads (8 waves).
// ---------------------------------------------------------------------------
__global__ __launch_bounds__(256) void edge_mlp_kernel(
    const float* __restrict__ edge_emb, const float* __restrict__ node_emb,
    const int* __restrict__ senders, const int* __restrict__ receivers,
    const float* __restrict__ dist,
    const __bf16* __restrict__ WT0, const float* __restrict__ B0,
    const __bf16* __restrict__ WT1, const float* __restrict__ B1,
    const __bf16* __restrict__ WT2, const float* __restrict__ B2,
    const float* __restrict__ lnG, const float* __restrict__ lnB,
    const float* __restrict__ rExp, const float* __restrict__ sExp,
    const float* __restrict__ rSum, const float* __restrict__ sSum,
    float* __restrict__ sentA, float* __restrict__ recvA,
    float* __restrict__ outEdge)
{
    __shared__ __attribute__((aligned(16))) char smem[SMEM_BYTES];
    __bf16* sEF  = (__bf16*)smem;               // 64x384, dead after layer0
    __bf16* sH0  = (__bf16*)(smem + 49152);     // 64x256
    __bf16* sH1  = (__bf16*)(smem + 81920);     // 64x256
    float*  sOut = (float*)smem;                // 64x128, aliases sEF

    const int e0 = blockIdx.x * MT;
    const int t = threadIdx.x;
    const int lane = t & 31, wave = t >> 5;

    // ---- gather [edge_emb | node_emb[s] | node_emb[r]] -> bf16 LDS -------
    {
        const int ts = t & 15;                  // 16 threads per row
        #pragma unroll
        for (int rb = 0; rb < 4; ++rb) {
            const int r = rb * 16 + (t >> 4);
            const int e = e0 + r;
            const int sN = senders[e], rN = receivers[e];
            const float* er = edge_emb + (size_t)e * CC;
            const float* ns = node_emb + (size_t)sN * CC;
            const float* nr = node_emb + (size_t)rN * CC;
            #pragma unroll
            for (int i = 0; i < 8; ++i) {
                const int c = ts + i * 16;
                sEF[r * K0 + c]           = (__bf16)er[c];
                sEF[r * K0 + CC + c]      = (__bf16)ns[c];
                sEF[r * K0 + 2 * CC + c]  = (__bf16)nr[c];
            }
        }
    }
    __syncthreads();

    mlp3_body(sEF, sH0, sH1, sOut, WT0, B0, WT1, B1, WT2, B2, lane, wave);

    // ---- layernorm + residual + weighted scatter (8 rows per wave) -------
    for (int rb = 0; rb < 8; ++rb) {
        const int rr = wave * 8 + rb;
        const int e = e0 + rr;
        const float4 v = *(const float4*)&sOut[rr * CC + lane * 4];
        float s1 = v.x + v.y + v.z + v.w;
        float s2 = v.x * v.x + v.y * v.y + v.z * v.z + v.w * v.w;
        #pragma unroll
        for (int off = 16; off > 0; off >>= 1) {
            s1 += __shfl_xor(s1, off, 32);
            s2 += __shfl_xor(s2, off, 32);
        }
        const float mean = s1 * (1.0f / CC);
        const float var  = s2 * (1.0f / CC) - mean * mean;
        const float rstd = rsqrtf(var + LN_EPS);

        const int sN = senders[e], rN = receivers[e];
        const float x = dist[e] * CUTOFF_INV;
        float env = 0.0f;
        if (x < 1.0f) {
            const float x2 = x * x, x4 = x2 * x2;
            env = 1.0f - 15.0f * x4 + 24.0f * x4 * x - 10.0f * x4 * x2;
        }
        const float sw = sExp[e] / sSum[sN] * env;
        const float rw = rExp[e] / rSum[rN] * env;

        const float xv[4] = {v.x, v.y, v.z, v.w};
        const float* ee = edge_emb + (size_t)e * CC;
        float* oe = outEdge + (size_t)e * CC;
        #pragma unroll
        for (int j = 0; j < 4; ++j) {
            const int c = lane * 4 + j;
            const float xn = (xv[j] - mean) * rstd * lnG[c] + lnB[c];
            oe[c] = ee[c] + xn;                                  // residual
            atomicAdd(&sentA[(size_t)sN * CC + c], xn * sw);     // scatter
            atomicAdd(&recvA[(size_t)rN * CC + c], xn * rw);
        }
    }
}

// ---------------------------------------------------------------------------
// Kernel 6: node MLP + layernorm + residual.  One block = 64 nodes.
// ---------------------------------------------------------------------------
__global__ __launch_bounds__(256) void node_mlp_kernel(
    const float* __restrict__ node_emb,
    const float* __restrict__ sentA, const float* __restrict__ recvA,
    const __bf16* __restrict__ WT0, const float* __restrict__ B0,
    const __bf16* __restrict__ WT1, const float* __restrict__ B1,
    const __bf16* __restrict__ WT2, const float* __restrict__ B2,
    const float* __restrict__ lnG, const float* __restrict__ lnB,
    float* __restrict__ outNode)
{
    __shared__ __attribute__((aligned(16))) char smem[SMEM_BYTES];
    __bf16* sEF  = (__bf16*)smem;
    __bf16* sH0  = (__bf16*)(smem + 49152);
    __bf16* sH1  = (__bf16*)(smem + 81920);
    float*  sOut = (float*)smem;

    const int n0 = blockIdx.x * MT;
    const int t = threadIdx.x;
    const int lane = t & 31, wave = t >> 5;

    {
        const int ts = t & 15;
        #pragma unroll
        for (int rb = 0; rb < 4; ++rb) {
            const int r = rb * 16 + (t >> 4);
            const int n = n0 + r;
            const float* a = node_emb + (size_t)n * CC;
            const float* b = sentA + (size_t)n * CC;
            const float* c = recvA + (size_t)n * CC;
            #pragma unroll
            for (int i = 0; i < 8; ++i) {
                const int cc = ts + i * 16;
                sEF[r * K0 + cc]          = (__bf16)a[cc];
                sEF[r * K0 + CC + cc]     = (__bf16)b[cc];
                sEF[r * K0 + 2 * CC + cc] = (__bf16)c[cc];
            }
        }
    }
    __syncthreads();

    mlp3_body(sEF, sH0, sH1, sOut, WT0, B0, WT1, B1, WT2, B2, lane, wave);

    for (int rb = 0; rb < 8; ++rb) {
        const int rr = wave * 8 + rb;
        const int n = n0 + rr;
        const float4 v = *(const float4*)&sOut[rr * CC + lane * 4];
        float s1 = v.x + v.y + v.z + v.w;
        float s2 = v.x * v.x + v.y * v.y + v.z * v.z + v.w * v.w;
        #pragma unroll
        for (int off = 16; off > 0; off >>= 1) {
            s1 += __shfl_xor(s1, off, 32);
            s2 += __shfl_xor(s2, off, 32);
        }
        const float mean = s1 * (1.0f / CC);
        const float var  = s2 * (1.0f / CC) - mean * mean;
        const float rstd = rsqrtf(var + LN_EPS);

        const float xv[4] = {v.x, v.y, v.z, v.w};
        const float* ne = node_emb + (size_t)n * CC;
        float* on = outNode + (size_t)n * CC;
        #pragma unroll
        for (int j = 0; j < 4; ++j) {
            const int c = lane * 4 + j;
            const float xn = (xv[j] - mean) * rstd * lnG[c] + lnB[c];
            on[c] = ne[c] + xn;
        }
    }
}

// ---------------------------------------------------------------------------
// Host-side launcher
// ---------------------------------------------------------------------------
extern "C" void kernel_launch(void* const* d_in, const int* in_sizes, int n_in,
                              void* d_out, int out_size, void* d_ws, size_t ws_size,
                              hipStream_t stream)
{
    const float* node_emb = (const float*)d_in[0];
    const float* edge_emb = (const float*)d_in[1];
    const int*   nl       = (const int*)d_in[2];
    const float* dist     = (const float*)d_in[3];
    const float* eW0 = (const float*)d_in[4];  const float* eB0 = (const float*)d_in[5];
    const float* eW1 = (const float*)d_in[6];  const float* eB1 = (const float*)d_in[7];
    const float* eW2 = (const float*)d_in[8];  const float* eB2 = (const float*)d_in[9];
    const float* eLnG = (const float*)d_in[10]; const float* eLnB = (const float*)d_in[11];
    const float* nW0 = (const float*)d_in[12]; const float* nB0 = (const float*)d_in[13];
    const float* nW1 = (const float*)d_in[14]; const float* nB1 = (const float*)d_in[15];
    const float* nW2 = (const float*)d_in[16]; const float* nB2 = (const float*)d_in[17];
    const float* nLnG = (const float*)d_in[18]; const float* nLnB = (const float*)d_in[19];
    const float* recv_w = (const float*)d_in[20]; const float* recv_b = (const float*)d_in[21];
    const float* send_w = (const float*)d_in[22]; const float* send_b = (const float*)d_in[23];

    const int* senders   = nl;          // neighbour_list[0]
    const int* receivers = nl + EE;     // neighbour_list[1]

    float* outNode = (float*)d_out;                   // [N, C]
    float* outEdge = outNode + (size_t)NN * CC;       // [E, C]

    // ---- workspace layout (f32 region then bf16 weights) -----------------
    float* ws   = (float*)d_ws;
    float* rLog = ws;                       // E   (becomes rExp)
    float* sLog = rLog + EE;                // E   (becomes sExp)
    float* rMax = sLog + EE;                // N
    float* sMax = rMax + NN;                // N
    float* rSum = sMax + NN;                // N
    float* sSum = rSum + NN;                // N
    float* sentA = sSum + NN;               // N*C
    float* recvA = sentA + (size_t)NN * CC; // N*C
    __bf16* wt   = (__bf16*)(recvA + (size_t)NN * CC);
    __bf16* wtE0 = wt;                       // 256x384
    __bf16* wtE1 = wtE0 + 384 * 256;         // 256x256
    __bf16* wtE2 = wtE1 + 256 * 256;         // 128x256
    __bf16* wtN0 = wtE2 + 256 * 128;
    __bf16* wtN1 = wtN0 + 384 * 256;
    __bf16* wtN2 = wtN1 + 256 * 256;

    // 1) init accumulators
    init_kernel<<<(NN * CC) / 256, 256, 0, stream>>>(rMax, sMax, rSum, sSum,
                                                     sentA, recvA);
    // 2) weight transpose -> bf16
    transpose_kernel<<<(384 * 256 + 255) / 256, 256, 0, stream>>>(eW0, wtE0, 384, 256);
    transpose_kernel<<<(256 * 256 + 255) / 256, 256, 0, stream>>>(eW1, wtE1, 256, 256);
    transpose_kernel<<<(256 * 128 + 255) / 256, 256, 0, stream>>>(eW2, wtE2, 256, 128);
    transpose_kernel<<<(384 * 256 + 255) / 256, 256, 0, stream>>>(nW0, wtN0, 384, 256);
    transpose_kernel<<<(256 * 256 + 255) / 256, 256, 0, stream>>>(nW1, wtN1, 256, 256);
    transpose_kernel<<<(256 * 128 + 255) / 256, 256, 0, stream>>>(nW2, wtN2, 256, 128);
    // 3) logits + segment max
    logits_kernel<<<EE / 256, 256, 0, stream>>>(edge_emb, recv_w, recv_b,
                                                send_w, send_b, senders,
                                                receivers, rLog, sLog, rMax, sMax);
    // 4) exp + segment sum
    expsum_kernel<<<EE / 256, 256, 0, stream>>>(senders, receivers, rLog, sLog,
                                                rMax, sMax, rSum, sSum);
    // 5) edge MLP + scatter
    edge_mlp_kernel<<<EE / MT, 256, 0, stream>>>(
        edge_emb, node_emb, senders, receivers, dist,
        wtE0, eB0, wtE1, eB1, wtE2, eB2, eLnG, eLnB,
        rLog, sLog, rSum, sSum, sentA, recvA, outEdge);
    // 6) node MLP
    node_mlp_kernel<<<NN / MT, 256, 0, stream>>>(
        node_emb, sentA, recvA,
        wtN0, nB0, wtN1, nB1, wtN2, nB2, nLnG, nLnB, outNode);
}
